// KTM_52080773431609
// MI455X (gfx1250) — matrix operand order repaired and checked by reading, not compiled, not run on hardware
//
#include <hip/hip_runtime.h>

// Problem constants (match reference): B=1024, F=4096, D=32
#define B_ROWS 1024
#define F_DIM  4096
#define D_DIM  32
#define N_COL  48                               // 32 embed cols + 1 bias col + 15 zero pad = 3 WMMA N-tiles
#define WAVES  16                               // waves per workgroup (wave32)
#define KSPLIT 4                                // K-blocks across grid.y
#define KCHUNK (F_DIM / (KSPLIT * WAVES))       // 64 -> 2 WMMA k-steps per wave

typedef __attribute__((ext_vector_type(16))) __bf16 v16bf;
typedef __attribute__((ext_vector_type(8)))  float  v8f;

union BFrag { v16bf v; uint4 q[2]; };

// ---------------------------------------------------------------------------
// Kernel 1: build transposed bf16 augmented B matrix  bT[48][4096]
//   c in [0,32): embed^T ; c == 32: bias ; c in (32,48): zeros
// ---------------------------------------------------------------------------
__global__ void KTM_prep_bT(const float* __restrict__ embed,
                            const float* __restrict__ bias,
                            __bf16* __restrict__ bT) {
  int tid = blockIdx.x * blockDim.x + threadIdx.x;
  if (tid >= N_COL * F_DIM) return;
  int c = tid >> 12;              // / 4096
  int k = tid & (F_DIM - 1);
  float v = 0.0f;
  if (c < D_DIM)       v = embed[k * D_DIM + c];
  else if (c == D_DIM) v = bias[k];
  bT[c * F_DIM + k] = (__bf16)v;
}

// ---------------------------------------------------------------------------
// Kernel 2: split-K GEMM partials via v_wmma_f32_16x16x32_bf16
//   grid = (64 row-tiles, KSPLIT k-blocks), block = 16 waves (512 threads)
//   part[kb][row][c] = partial s_aug for that K-block
// ---------------------------------------------------------------------------
__global__ __launch_bounds__(WAVES * 32)
void KTM_fm_gemm_part(const float* __restrict__ data,
                      const __bf16* __restrict__ bT,
                      float* __restrict__ part) {
  __shared__ float lds[WAVES][16][N_COL];       // 48 KB

  const int tid  = threadIdx.x;
  const int wave = tid >> 5;
  const int lane = tid & 31;
  const int r    = lane & 15;                   // A: row within tile; B: column within N-tile
  const int hi   = lane >> 4;                   // half-wave selector
  const int rowBase = blockIdx.x * 16;
  const int kBlk    = blockIdx.y;

  v8f acc0 = {}, acc1 = {}, acc2 = {};

  const int kStart = (kBlk * WAVES + wave) * KCHUNK;
  const float* arow = data + (size_t)(rowBase + r) * F_DIM;

  #pragma unroll
  for (int kk = 0; kk < KCHUNK; kk += 32) {
    const int k0 = kStart + kk;

    // ---- A fragment (16x32 fp32 -> bf16), documented 16-bit A layout:
    //      lanes 0-15: K = 0..7,16..23 ; lanes 16-31: K = 8..15,24..31
    const float* ap = arow + k0 + hi * 8;
    float4 a0 = *(const float4*)(ap + 0);
    float4 a1 = *(const float4*)(ap + 4);
    float4 a2 = *(const float4*)(ap + 16);
    float4 a3 = *(const float4*)(ap + 20);
    v16bf av;
    av[0]  = (__bf16)a0.x; av[1]  = (__bf16)a0.y; av[2]  = (__bf16)a0.z; av[3]  = (__bf16)a0.w;
    av[4]  = (__bf16)a1.x; av[5]  = (__bf16)a1.y; av[6]  = (__bf16)a1.z; av[7]  = (__bf16)a1.w;
    av[8]  = (__bf16)a2.x; av[9]  = (__bf16)a2.y; av[10] = (__bf16)a2.z; av[11] = (__bf16)a2.w;
    av[12] = (__bf16)a3.x; av[13] = (__bf16)a3.y; av[14] = (__bf16)a3.z; av[15] = (__bf16)a3.w;

    // ---- B fragments (32x16 bf16), 16-bit B layout:
    //      lanes 0-15 hold K=0..7,16..23, lanes 16-31 hold K=8..15,24..31, lane%16 = N.
    //      bT is K-major -> each lane is two aligned 128-bit loads per N-tile.
    const __bf16* bp = bT + k0 + hi * 8;
    BFrag b0, b1, b2;
    b0.q[0] = *(const uint4*)(bp + (0 * 16 + r) * F_DIM);
    b0.q[1] = *(const uint4*)(bp + (0 * 16 + r) * F_DIM + 16);
    b1.q[0] = *(const uint4*)(bp + (1 * 16 + r) * F_DIM);
    b1.q[1] = *(const uint4*)(bp + (1 * 16 + r) * F_DIM + 16);
    b2.q[0] = *(const uint4*)(bp + (2 * 16 + r) * F_DIM);
    b2.q[1] = *(const uint4*)(bp + (2 * 16 + r) * F_DIM + 16);

    acc0 = __builtin_amdgcn_wmma_f32_16x16x32_bf16(false, av, false, b0.v, (short)0, acc0, false, false);
    acc1 = __builtin_amdgcn_wmma_f32_16x16x32_bf16(false, av, false, b1.v, (short)0, acc1, false, false);
    acc2 = __builtin_amdgcn_wmma_f32_16x16x32_bf16(false, av, false, b2.v, (short)0, acc2, false, false);
  }

  // ---- spill accumulator fragments to LDS (C/D layout: lanes 0-15 -> M=i, lanes 16-31 -> M=8+i)
  #pragma unroll
  for (int i = 0; i < 8; ++i) {
    const int m = hi * 8 + i;
    lds[wave][m][0 * 16 + r] = acc0[i];
    lds[wave][m][1 * 16 + r] = acc1[i];
    lds[wave][m][2 * 16 + r] = acc2[i];
  }
  __syncthreads();

  // ---- deterministic fixed-order reduction over the 16 waves, write partial tile
  float* pout = part + ((size_t)kBlk * B_ROWS + rowBase) * N_COL;
  for (int p = tid; p < 16 * N_COL; p += WAVES * 32) {
    const int m = p / N_COL;
    const int c = p - m * N_COL;
    float s = 0.0f;
    #pragma unroll
    for (int w = 0; w < WAVES; ++w) s += lds[w][m][c];
    pout[m * N_COL + c] = s;
  }
}

// ---------------------------------------------------------------------------
// Kernel 3: combine K-block partials, ||s||^2 + bias + gb, sigmoid
// ---------------------------------------------------------------------------
__global__ void KTM_fm_finalize(const float* __restrict__ part,
                                const float* __restrict__ gbias,
                                float* __restrict__ out) {
  int row = blockIdx.x * blockDim.x + threadIdx.x;
  if (row >= B_ROWS) return;

  float sv[D_DIM + 1];
  #pragma unroll
  for (int c = 0; c <= D_DIM; ++c) sv[c] = 0.0f;

  #pragma unroll
  for (int kb = 0; kb < KSPLIT; ++kb) {
    const float* p = part + ((size_t)kb * B_ROWS + row) * N_COL;
    #pragma unroll
    for (int c = 0; c <= D_DIM; ++c) sv[c] += p[c];
  }

  float dotv = 0.0f;
  #pragma unroll
  for (int c = 0; c < D_DIM; ++c) dotv += sv[c] * sv[c];

  float x = gbias[0] + sv[D_DIM] + dotv;
  out[row] = 1.0f / (1.0f + __expf(-x));
}

// ---------------------------------------------------------------------------
extern "C" void kernel_launch(void* const* d_in, const int* in_sizes, int n_in,
                              void* d_out, int out_size, void* d_ws, size_t ws_size,
                              hipStream_t stream) {
  (void)in_sizes; (void)n_in; (void)out_size; (void)ws_size;

  const float* data  = (const float*)d_in[0];   // (B,F) f32
  const float* embed = (const float*)d_in[1];   // (F,D) f32
  const float* bias  = (const float*)d_in[2];   // (F,1) f32
  const float* gb    = (const float*)d_in[3];   // (1,1) f32
  float*       out   = (float*)d_out;           // (B,)  f32

  // workspace layout: [bT: 48*4096 bf16 = 384 KB][part: KSPLIT*1024*48 f32 = 768 KB]
  __bf16* bT   = (__bf16*)d_ws;
  float*  part = (float*)((char*)d_ws + (size_t)N_COL * F_DIM * sizeof(__bf16));

  int prepN = N_COL * F_DIM;
  KTM_prep_bT<<<(prepN + 255) / 256, 256, 0, stream>>>(embed, bias, bT);

  dim3 grid(B_ROWS / 16, KSPLIT);
  KTM_fm_gemm_part<<<grid, WAVES * 32, 0, stream>>>(data, bT, part);

  KTM_fm_finalize<<<(B_ROWS + 255) / 256, 256, 0, stream>>>(part, gb, out);
}
